// Q_JetTagger_32839319945217
// MI455X (gfx1250) — 8-bit matrix operand order repaired and checked by reading, compiled, not run on hardware
//
#include <hip/hip_runtime.h>
#include <hip/hip_bf16.h>
#include <stdint.h>

typedef __attribute__((ext_vector_type(8))) int v8i;

#define WS_M        0
#define WS_CMAX     16
#define WS_COLSUM   64
#define WS_SCALEF   84
#define WS_INVSF    88
#define WS_ACT      96
#define WS_INVACT   112
#define WS_WQ1      1024
#define WS_BI1      5120
#define WS_BS1      5376
#define WS_WQ2      5632
#define WS_BI2      7680
#define WS_BS2      7808
#define WS_WQ3      7936
#define WS_BI3      9984
#define WS_BS3      10112
#define WS_WQ4      10240
#define WS_BI4      11264
#define WS_BS4      11328
#define WS_PART     12288
#define WS_H4       32768
#define NSUM_BLOCKS 512

__device__ __forceinline__ unsigned flipEnc(float x) {
  unsigned u = __float_as_uint(x);
  return (u & 0x80000000u) ? ~u : (u | 0x80000000u);
}
__device__ __forceinline__ float flipDec(unsigned u) {
  return __uint_as_float((u & 0x80000000u) ? (u & 0x7fffffffu) : ~u);
}

__device__ __forceinline__ v8i ldsAfrag(const int* plane, int m0, int lane) {
  int row = m0 + (lane & 15);
  int b = row * 16 + ((lane & 16) ? 2 : 0);
  v8i f;
  f[0] = plane[b + 0];  f[1] = plane[b + 1];
  f[2] = plane[b + 4];  f[3] = plane[b + 5];
  f[4] = plane[b + 8];  f[5] = plane[b + 9];
  f[6] = plane[b + 12]; f[7] = plane[b + 13];
  return f;
}
__device__ __forceinline__ v8i ldsBfrag(const int* w, int colBase, int lane) {
  int col = colBase + (lane & 15);
  int b = col * 16 + ((lane & 16) ? 2 : 0);
  v8i f;
  f[0] = w[b + 0]; f[1] = w[b + 1]; f[2] = w[b + 4];  f[3] = w[b + 5];
  f[4] = w[b + 8]; f[5] = w[b + 9]; f[6] = w[b + 12]; f[7] = w[b + 13];
  return f;
}

__global__ void k_init(char* ws) {
  if (threadIdx.x < 16) ((unsigned*)ws)[threadIdx.x] = 0u;
}

__global__ __launch_bounds__(256) void k_absmax(const float4* __restrict__ x,
                                                char* __restrict__ ws, int n4) {
  float m = 0.f;
  for (int i = blockIdx.x * 256 + threadIdx.x; i < n4; i += gridDim.x * 256) {
    float4 v = x[i];
    m = fmaxf(m, fmaxf(fmaxf(fabsf(v.x), fabsf(v.y)), fmaxf(fabsf(v.z), fabsf(v.w))));
  }
  __shared__ float red[256];
  red[threadIdx.x] = m;
  __syncthreads();
  for (int off = 128; off > 0; off >>= 1) {
    if (threadIdx.x < off) red[threadIdx.x] = fmaxf(red[threadIdx.x], red[threadIdx.x + off]);
    __syncthreads();
  }
  if (threadIdx.x == 0) atomicMax((unsigned*)(ws + WS_M), __float_as_uint(red[0]));
}

__global__ void k_prep(const float* __restrict__ W, const float* __restrict__ b,
                       int Kin, int Nreal, int Npad, int layer, char* __restrict__ ws) {
  const int wqo[4] = {WS_WQ1, WS_WQ2, WS_WQ3, WS_WQ4};
  const int bio[4] = {WS_BI1, WS_BI2, WS_BI3, WS_BI4};
  const int bso[4] = {WS_BS1, WS_BS2, WS_BS3, WS_BS4};
  int8_t* wq = (int8_t*)(ws + wqo[layer]);
  int* bint = (int*)(ws + bio[layer]);
  float* bsc = (float*)(ws + bso[layer]);
  __shared__ float sSc;
  if (threadIdx.x == 0) {
    float sat = __uint_as_float(((const unsigned*)(ws + WS_M))[layer]);
    float sc = fmaxf(sat, 1e-8f) / 32767.f;
    ((float*)(ws + WS_ACT))[layer] = sc;
    ((float*)(ws + WS_INVACT))[layer] = 1.f / sc;
    sSc = sc;
  }
  __syncthreads();
  int o = threadIdx.x;
  if (o < Npad) {
    if (o < Nreal) {
      float m = 0.f;
      for (int k = 0; k < Kin; ++k) m = fmaxf(m, fabsf(W[o * Kin + k]));
      float wsc = fmaxf(m, 1e-8f) / 127.f;
      for (int k = 0; k < 64; ++k) {
        int q = 0;
        if (k < Kin) q = (int)fminf(fmaxf(rintf(W[o * Kin + k] / wsc), -128.f), 127.f);
        wq[o * 64 + k] = (int8_t)q;
      }
      float bscale = wsc * sSc;
      double r = rint((double)b[o] / (double)bscale);
      r = fmin(fmax(r, -2147483648.0), 2147483647.0);
      bint[o] = (int)(long long)r;
      bsc[o] = bscale;
    } else {
      for (int k = 0; k < 64; ++k) wq[o * 64 + k] = 0;
      bint[o] = 0;
      bsc[o] = 0.f;
    }
  }
}

template <int DEPTH>
__global__ __launch_bounds__(256) void k_chain(const float* __restrict__ x,
                                               char* __restrict__ ws) {
  constexpr int NP[4]  = {64, 32, 32, 16};
  constexpr int WB[4]  = {0, 4096, 6144, 8192};
  constexpr int WQO[4] = {WS_WQ1, WS_WQ2, WS_WQ3, WS_WQ4};
  constexpr int BIO[4] = {WS_BI1, WS_BI2, WS_BI3, WS_BI4};
  constexpr int BSO[4] = {WS_BS1, WS_BS2, WS_BS3, WS_BS4};

  __shared__ int planesHi[2][2048];
  __shared__ int planesLo[2][2048];
  __shared__ int wS[2304];
  __shared__ float sred[256];
  __shared__ unsigned cred[5];

  const int tid = threadIdx.x;
  const int lane = tid & 31;
  const int wave = tid >> 5;
  const int row0 = blockIdx.x * 128;

#pragma unroll
  for (int s = 0; s < DEPTH; ++s) {
    const int* src = (const int*)(ws + WQO[s]);
    const int nInts = NP[s] * 16;
    for (int i = tid; i < nInts; i += 256) wS[(WB[s] >> 2) + i] = src[i];
  }
  if (DEPTH == 4 && tid < 5) cred[tid] = 0u;

  {
    const float inv0 = ((const float*)(ws + WS_INVACT))[0];
    char* pH = (char*)planesHi[0];
    char* pL = (char*)planesLo[0];
    for (int i = tid; i < 128 * 16; i += 256) {
      float v = x[row0 * 16 + i];
      int qi = (int)fminf(fmaxf(rintf(v * inv0), -32768.f), 32767.f);
      int r = i >> 4, k = i & 15;
      pH[r * 64 + k] = (char)(qi >> 8);
      pL[r * 64 + k] = (char)(qi & 0xff);
    }
  }

  float localMax = 0.0f;
  float cMax = -3.4e38f;
  float* h4 = (float*)(ws + WS_H4);

#pragma unroll
  for (int s = 0; s < DEPTH; ++s) {
    __syncthreads();
    const int* inHi = planesHi[s & 1];
    const int* inLo = planesLo[s & 1];
    char* outHi = (char*)planesHi[(s + 1) & 1];
    char* outLo = (char*)planesLo[(s + 1) & 1];
    const int m0 = wave * 16;
    v8i aHi = ldsAfrag(inHi, m0, lane);
    v8i aLo = ldsAfrag(inLo, m0, lane);
    const int* bint = (const int*)(ws + BIO[s]);
    const float* bscale = (const float*)(ws + BSO[s]);
    const float invNext = (s + 1 < DEPTH) ? ((const float*)(ws + WS_INVACT))[s + 1] : 0.f;

#pragma unroll
    for (int t = 0; t < NP[s] / 16; ++t) {
      v8i bF = ldsBfrag(wS + (WB[s] >> 2), t * 16, lane);
      v8i accH = {0, 0, 0, 0, 0, 0, 0, 0};
      v8i accL = {0, 0, 0, 0, 0, 0, 0, 0};
      accH = __builtin_amdgcn_wmma_i32_16x16x64_iu8(true, aHi, true, bF, accH, false, false);
      accL = __builtin_amdgcn_wmma_i32_16x16x64_iu8(false, aLo, true, bF, accL, false, false);
      const int col = t * 16 + (lane & 15);
      const int bi = bint[col];
      const float bs = bscale[col];
      const int rbase = m0 + ((lane >> 4) << 3);
#pragma unroll
      for (int e = 0; e < 8; ++e) {
        int acc = accH[e] * 256 + accL[e] + bi;
        float val = (float)acc * bs;
        const int rl = rbase + e;
        if (s + 1 < DEPTH) {
          val = fmaxf(val, 0.f);
          int q = (int)fminf(fmaxf(rintf(val * invNext), -32768.f), 32767.f);
          outHi[rl * 64 + col] = (char)(q >> 8);
          outLo[rl * 64 + col] = (char)(q & 0xff);
        } else if (DEPTH < 4) {
          val = fmaxf(val, 0.f);
          localMax = fmaxf(localMax, val);
        } else {
          if (col < 5) {
            h4[(row0 + rl) * 5 + col] = val;
            cMax = fmaxf(cMax, val);
          }
        }
      }
    }
  }

  if (DEPTH < 4) {
    __syncthreads();
    sred[tid] = localMax;
    __syncthreads();
    for (int off = 128; off > 0; off >>= 1) {
      if (tid < off) sred[tid] = fmaxf(sred[tid], sred[tid + off]);
      __syncthreads();
    }
    if (tid == 0)
      atomicMax(((unsigned*)(ws + WS_M)) + DEPTH, __float_as_uint(sred[0]));
  } else {
    const int col = lane & 15;
    if (col < 5) atomicMax(&cred[col], flipEnc(cMax));
    __syncthreads();
    if (tid < 5) atomicMax(((unsigned*)(ws + WS_CMAX)) + tid, cred[tid]);
  }
}

__global__ __launch_bounds__(256) void k_expsum(char* __restrict__ ws, int B) {
  const float* h4 = (const float*)(ws + WS_H4);
  const unsigned* cmb = (const unsigned*)(ws + WS_CMAX);
  float cm[5];
#pragma unroll
  for (int j = 0; j < 5; ++j) cm[j] = flipDec(cmb[j]);
  float loc[5] = {0.f, 0.f, 0.f, 0.f, 0.f};
  for (int r = blockIdx.x * 256 + threadIdx.x; r < B; r += gridDim.x * 256) {
#pragma unroll
    for (int j = 0; j < 5; ++j) loc[j] += expf(h4[r * 5 + j] - cm[j]);
  }
  __shared__ float red[256];
  float* part = (float*)(ws + WS_PART);
#pragma unroll
  for (int j = 0; j < 5; ++j) {
    red[threadIdx.x] = loc[j];
    __syncthreads();
    for (int off = 128; off > 0; off >>= 1) {
      if (threadIdx.x < off) red[threadIdx.x] += red[threadIdx.x + off];
      __syncthreads();
    }
    if (threadIdx.x == 0) part[blockIdx.x * 5 + j] = red[0];
    __syncthreads();
  }
}

__global__ void k_expfinal(char* __restrict__ ws) {
  __shared__ float cs[5];
  int j = threadIdx.x;
  if (j < 5) {
    const float* part = (const float*)(ws + WS_PART);
    float s = 0.f;
    for (int b = 0; b < NSUM_BLOCKS; ++b) s += part[b * 5 + j];
    ((float*)(ws + WS_COLSUM))[j] = s;
    cs[j] = s;
  }
  __syncthreads();
  if (j == 0) {
    float sat = 0.f;
    for (int t = 0; t < 5; ++t) sat = fmaxf(sat, 1.f / cs[t]);
    float sc = fmaxf(sat, 1e-8f) / 32767.f;
    *(float*)(ws + WS_SCALEF) = sc;
    *(float*)(ws + WS_INVSF) = 1.f / sc;
  }
}

__global__ __launch_bounds__(256) void k_out(const char* __restrict__ ws,
                                             float* __restrict__ out, int B) {
  const float* h4 = (const float*)(ws + WS_H4);
  const unsigned* cmb = (const unsigned*)(ws + WS_CMAX);
  const float* cs = (const float*)(ws + WS_COLSUM);
  float cm[5], rs[5];
#pragma unroll
  for (int j = 0; j < 5; ++j) {
    cm[j] = flipDec(cmb[j]);
    rs[j] = 1.f / cs[j];
  }
  const float sc = *(const float*)(ws + WS_SCALEF);
  const float inv = *(const float*)(ws + WS_INVSF);
  for (int r = blockIdx.x * 256 + threadIdx.x; r < B; r += gridDim.x * 256) {
#pragma unroll
    for (int j = 0; j < 5; ++j) {
      float v = expf(h4[r * 5 + j] - cm[j]) * rs[j];
      float q = fminf(fmaxf(rintf(v * inv), -32768.f), 32767.f);
      out[r * 5 + j] = q * sc;
    }
  }
}

extern "C" void kernel_launch(void* const* d_in, const int* in_sizes, int n_in,
                              void* d_out, int out_size, void* d_ws, size_t ws_size,
                              hipStream_t stream) {
  (void)n_in; (void)out_size; (void)ws_size;
  const float* x  = (const float*)d_in[0];
  const float* W1 = (const float*)d_in[1];
  const float* b1 = (const float*)d_in[2];
  const float* W2 = (const float*)d_in[3];
  const float* b2 = (const float*)d_in[4];
  const float* W3 = (const float*)d_in[5];
  const float* b3 = (const float*)d_in[6];
  const float* W4 = (const float*)d_in[7];
  const float* b4 = (const float*)d_in[8];
  char* ws = (char*)d_ws;
  const int B = in_sizes[0] / 16;
  const int nblk = B / 128;

  hipLaunchKernelGGL(k_init, dim3(1), dim3(32), 0, stream, ws);
  hipLaunchKernelGGL(k_absmax, dim3(2048), dim3(256), 0, stream,
                     (const float4*)x, ws, B * 4);
  hipLaunchKernelGGL(k_prep, dim3(1), dim3(64), 0, stream, W1, b1, 16, 64, 64, 0, ws);
  hipLaunchKernelGGL(k_chain<1>, dim3(nblk), dim3(256), 0, stream, x, ws);
  hipLaunchKernelGGL(k_prep, dim3(1), dim3(64), 0, stream, W2, b2, 64, 32, 32, 1, ws);
  hipLaunchKernelGGL(k_chain<2>, dim3(nblk), dim3(256), 0, stream, x, ws);
  hipLaunchKernelGGL(k_prep, dim3(1), dim3(64), 0, stream, W3, b3, 32, 32, 32, 2, ws);
  hipLaunchKernelGGL(k_chain<3>, dim3(nblk), dim3(256), 0, stream, x, ws);
  hipLaunchKernelGGL(k_prep, dim3(1), dim3(64), 0, stream, W4, b4, 32, 5, 16, 3, ws);
  hipLaunchKernelGGL(k_chain<4>, dim3(nblk), dim3(256), 0, stream, x, ws);
  hipLaunchKernelGGL(k_expsum, dim3(NSUM_BLOCKS), dim3(256), 0, stream, ws, B);
  hipLaunchKernelGGL(k_expfinal, dim3(1), dim3(32), 0, stream, ws);
  hipLaunchKernelGGL(k_out, dim3(2048), dim3(256), 0, stream, ws, (float*)d_out, B);
}
